// DGI_54176717471773
// MI455X (gfx1250) — matrix operand order repaired and checked by reading, not compiled, not run on hardware
//
#include <hip/hip_runtime.h>
#include <hip/hip_bf16.h>

typedef __attribute__((ext_vector_type(16))) _Float16 v16h;
typedef __attribute__((ext_vector_type(8)))  float    v8f;

#define N_NODES 50000
#define N_EDGES 500000

// ---------------------------------------------------------------------------
// helpers
// ---------------------------------------------------------------------------
__device__ __forceinline__ v8f wmma16(v16h a, v16h b, v8f c) {
    return __builtin_amdgcn_wmma_f32_16x16x32_f16(
        /*neg_a=*/false, a, /*neg_b=*/false, b,
        /*c_mod=*/(short)0, c, /*reuse_a=*/false, /*reuse_b=*/false);
}

// Build the per-lane A fragment for V_WMMA_F32_16X16X32_F16.
// halves 0..7  -> cols c0..c0+7 of this row
// halves 8..15 -> cols c0+16..c0+23
// (c0 already includes the +8 offset for lanes 16..31)
__device__ __forceinline__ v16h load_a_frag(const float* __restrict__ rowptr,
                                            int c0, float scale) {
    const float4* p = reinterpret_cast<const float4*>(rowptr) + (c0 >> 2);
    float4 x0 = p[0], x1 = p[1], x2 = p[4], x3 = p[5];
    v16h a;
    a[0] = (_Float16)(x0.x * scale);  a[1] = (_Float16)(x0.y * scale);
    a[2] = (_Float16)(x0.z * scale);  a[3] = (_Float16)(x0.w * scale);
    a[4] = (_Float16)(x1.x * scale);  a[5] = (_Float16)(x1.y * scale);
    a[6] = (_Float16)(x1.z * scale);  a[7] = (_Float16)(x1.w * scale);
    a[8] = (_Float16)(x2.x * scale);  a[9] = (_Float16)(x2.y * scale);
    a[10] = (_Float16)(x2.z * scale); a[11] = (_Float16)(x2.w * scale);
    a[12] = (_Float16)(x3.x * scale); a[13] = (_Float16)(x3.y * scale);
    a[14] = (_Float16)(x3.z * scale); a[15] = (_Float16)(x3.w * scale);
    return a;
}

__device__ __forceinline__ float softplus_f(float x) {
    return fmaxf(x, 0.0f) + log1pf(expf(-fabsf(x)));
}

// ---------------------------------------------------------------------------
// small utility kernels
// ---------------------------------------------------------------------------
__global__ void zero_f32_kernel(float* __restrict__ p, int n) {
    int i = blockIdx.x * blockDim.x + threadIdx.x;
    if (i < n) p[i] = 0.0f;
}

__global__ void deg_count_kernel(const int* __restrict__ dst, float* __restrict__ cnt, int E) {
    int e = blockIdx.x * blockDim.x + threadIdx.x;
    if (e < E) atomicAdd(cnt + dst[e], 1.0f);
}

__global__ void invdeg_kernel(float* __restrict__ cnt, int N) {
    int i = blockIdx.x * blockDim.x + threadIdx.x;
    if (i < N) cnt[i] = 1.0f / fmaxf(cnt[i], 1.0f);
}

// scatter-add message m_e = [node_h[src[e]] (64) | edge_f[erow] (64)] into agg[dst[e], 0:128]
__global__ void scatter_msgs_kernel(const float* __restrict__ node_h,
                                    const float* __restrict__ edge_f,
                                    const int* __restrict__ src,
                                    const int* __restrict__ dst,
                                    const int* __restrict__ perm, int use_perm,
                                    float* __restrict__ agg, int E) {
    int tid = blockIdx.x * blockDim.x + threadIdx.x;   // E*32 threads
    if (tid >= E * 32) return;
    int e  = tid >> 5;
    int c0 = (tid & 31) * 4;
    float4 v;
    if (c0 < 64) {
        v = *reinterpret_cast<const float4*>(node_h + (long)src[e] * 64 + c0);
    } else {
        int er = use_perm ? perm[e] : e;
        v = *reinterpret_cast<const float4*>(edge_f + (long)er * 64 + (c0 - 64));
    }
    float* ap = agg + (long)dst[e] * 128 + c0;
    atomicAdd(ap + 0, v.x); atomicAdd(ap + 1, v.y);
    atomicAdd(ap + 2, v.z); atomicAdd(ap + 3, v.w);
}

// ---------------------------------------------------------------------------
// WMMA GEMM: h_out = relu([h_in | agg*invdeg] @ W^T + b)   rows x 192 @ 192 x 64
// ---------------------------------------------------------------------------
__global__ void gemm_apply_kernel(const float* __restrict__ hin,     // [rows,64]
                                  const float* __restrict__ agg,     // [rows,128]
                                  const float* __restrict__ invdeg,  // [rows]
                                  const float* __restrict__ W,       // [64,192]
                                  const float* __restrict__ bias,    // [64]
                                  float* __restrict__ out,           // [rows,64]
                                  int rows) {
    __shared__ __align__(32) _Float16 wl[192 * 64];   // W[o][k]
    for (int i = threadIdx.x; i < 192 * 64; i += blockDim.x) {
        int k = i >> 6, o = i & 63;
        wl[o * 192 + k] = (_Float16)W[o * 192 + k];
    }
    __syncthreads();

    int wave = threadIdx.x >> 5;
    int lane = threadIdx.x & 31;
    int mbase = blockIdx.x * 128 + wave * 16;
    if (mbase >= rows) return;                         // wave-uniform

    int r = mbase + (lane & 15);
    if (r >= rows) r = rows - 1;
    float s   = invdeg[r];
    int  koff = (lane >= 16) ? 8 : 0;

    v8f acc0 = {}, acc1 = {}, acc2 = {}, acc3 = {};
#pragma unroll
    for (int kc = 0; kc < 6; ++kc) {
        int kb = kc * 32;
        v16h a;
        if (kb < 64)
            a = load_a_frag(hin + (long)r * 64, kb + koff, 1.0f);
        else
            a = load_a_frag(agg + (long)r * 128, (kb - 64) + koff, s);
        const _Float16* bp = &wl[(lane & 15) * 192 + kb + koff];
        auto ldb = [&](int j) { const _Float16* q = bp + j * 16 * 192; v16h b; for (int t = 0; t < 8; ++t) { b[t] = q[t]; b[t + 8] = q[t + 16]; } return b; };
        acc0 = wmma16(a, ldb(0), acc0);
        acc1 = wmma16(a, ldb(1), acc1);
        acc2 = wmma16(a, ldb(2), acc2);
        acc3 = wmma16(a, ldb(3), acc3);
    }

    int nb = lane & 15;
    float b0 = bias[nb], b1 = bias[16 + nb], b2 = bias[32 + nb], b3 = bias[48 + nb];
    int mo = (lane >= 16) ? 8 : 0;
#pragma unroll
    for (int j = 0; j < 8; ++j) {
        int m = mbase + mo + j;
        if (m < rows) {
            float* op = out + (long)m * 64 + nb;
            float v0 = acc0[j] + b0; op[0]  = v0 > 0.0f ? v0 : 0.0f;
            float v1 = acc1[j] + b1; op[16] = v1 > 0.0f ? v1 : 0.0f;
            float v2 = acc2[j] + b2; op[32] = v2 > 0.0f ? v2 : 0.0f;
            float v3 = acc3[j] + b3; op[48] = v3 > 0.0f ? v3 : 0.0f;
        }
    }
}

// ---------------------------------------------------------------------------
// WMMA GEMM: e_out = relu([h[src[e]] | h[dst[e]]] @ W^T + b)   E x 128 @ 128 x 64
// ---------------------------------------------------------------------------
__global__ void gemm_edge_kernel(const float* __restrict__ h,    // [N,64]
                                 const int* __restrict__ src,
                                 const int* __restrict__ dst,
                                 const float* __restrict__ W,    // [64,128]
                                 const float* __restrict__ bias, // [64]
                                 float* __restrict__ out,        // [E,64]
                                 int E) {
    __shared__ __align__(32) _Float16 wl[128 * 64];   // W[o][k]
    for (int i = threadIdx.x; i < 128 * 64; i += blockDim.x) {
        int k = i >> 6, o = i & 63;
        wl[o * 128 + k] = (_Float16)W[o * 128 + k];
    }
    __syncthreads();

    int wave = threadIdx.x >> 5;
    int lane = threadIdx.x & 31;
    int mbase = blockIdx.x * 128 + wave * 16;
    if (mbase >= E) return;                            // wave-uniform

    int e = mbase + (lane & 15);
    if (e >= E) e = E - 1;
    int rs = src[e], rd = dst[e];
    int koff = (lane >= 16) ? 8 : 0;

    v8f acc0 = {}, acc1 = {}, acc2 = {}, acc3 = {};
#pragma unroll
    for (int kc = 0; kc < 4; ++kc) {
        int kb = kc * 32;
        int row = (kb < 64) ? rs : rd;
        v16h a = load_a_frag(h + (long)row * 64, (kb & 63) + koff, 1.0f);
        const _Float16* bp = &wl[(lane & 15) * 128 + kb + koff];
        auto ldb = [&](int j) { const _Float16* q = bp + j * 16 * 128; v16h b; for (int t = 0; t < 8; ++t) { b[t] = q[t]; b[t + 8] = q[t + 16]; } return b; };
        acc0 = wmma16(a, ldb(0), acc0);
        acc1 = wmma16(a, ldb(1), acc1);
        acc2 = wmma16(a, ldb(2), acc2);
        acc3 = wmma16(a, ldb(3), acc3);
    }

    int nb = lane & 15;
    float b0 = bias[nb], b1 = bias[16 + nb], b2 = bias[32 + nb], b3 = bias[48 + nb];
    int mo = (lane >= 16) ? 8 : 0;
#pragma unroll
    for (int j = 0; j < 8; ++j) {
        int m = mbase + mo + j;
        if (m < E) {
            float* op = out + (long)m * 64 + nb;
            float v0 = acc0[j] + b0; op[0]  = v0 > 0.0f ? v0 : 0.0f;
            float v1 = acc1[j] + b1; op[16] = v1 > 0.0f ? v1 : 0.0f;
            float v2 = acc2[j] + b2; op[32] = v2 > 0.0f ? v2 : 0.0f;
            float v3 = acc3[j] + b3; op[48] = v3 > 0.0f ? v3 : 0.0f;
        }
    }
}

// ---------------------------------------------------------------------------
// readout kernels
// ---------------------------------------------------------------------------
__global__ void colsum_kernel(const float* __restrict__ pos, float* __restrict__ colsum, int E) {
    int col = threadIdx.x & 63;
    int rl  = threadIdx.x >> 6;            // 0..3
    float s = 0.0f;
    for (int r = blockIdx.x * 4 + rl; r < E; r += gridDim.x * 4)
        s += pos[(long)r * 64 + col];
    atomicAdd(colsum + col, s);
}

// summary = sigmoid(colsum/E);  t = W_bil @ summary;  zero loss accumulator
__global__ void make_t_kernel(const float* __restrict__ colsum,
                              const float* __restrict__ Wbil,
                              float* __restrict__ t, float* __restrict__ lossacc,
                              float invE) {
    __shared__ float s[64];
    int i = threadIdx.x;                    // 64 threads
    s[i] = 1.0f / (1.0f + expf(-colsum[i] * invE));
    __syncthreads();
    float acc = 0.0f;
    for (int f = 0; f < 64; ++f) acc += Wbil[i * 64 + f] * s[f];
    t[i] = acc;
    if (i == 0) lossacc[0] = 0.0f;
}

__global__ void loss_kernel(const float* __restrict__ pos, const float* __restrict__ neg,
                            const float* __restrict__ t, const float* __restrict__ bbil,
                            float* __restrict__ acc, int E) {
    __shared__ float tl[64];
    __shared__ float bsum;
    if (threadIdx.x < 64) tl[threadIdx.x] = t[threadIdx.x];
    if (threadIdx.x == 0) bsum = 0.0f;
    __syncthreads();
    int e = blockIdx.x * blockDim.x + threadIdx.x;
    float local = 0.0f;
    if (e < E) {
        float b = bbil[0];
        float sp = b, sn = b;
        const float* pp = pos + (long)e * 64;
        const float* np = neg + (long)e * 64;
#pragma unroll 8
        for (int d = 0; d < 64; ++d) { sp += pp[d] * tl[d]; sn += np[d] * tl[d]; }
        local = softplus_f(-sp) + softplus_f(sn);
    }
    atomicAdd(&bsum, local);
    __syncthreads();
    if (threadIdx.x == 0) atomicAdd(acc, bsum);
}

__global__ void finalize_kernel(const float* __restrict__ acc, float* __restrict__ out, float invE) {
    if (threadIdx.x == 0 && blockIdx.x == 0) out[0] = acc[0] * invE;
}

// ---------------------------------------------------------------------------
// launch
// ---------------------------------------------------------------------------
extern "C" void kernel_launch(void* const* d_in, const int* in_sizes, int n_in,
                              void* d_out, int out_size, void* d_ws, size_t ws_size,
                              hipStream_t stream) {
    (void)in_sizes; (void)n_in; (void)out_size; (void)ws_size;

    const int N = N_NODES, E = N_EDGES;

    const float* nfeats = (const float*)d_in[0];
    const float* efeats = (const float*)d_in[1];
    const int*   src    = (const int*)d_in[2];
    const int*   dst    = (const int*)d_in[3];
    const int*   perm   = (const int*)d_in[4];
    const float* Wa1    = (const float*)d_in[5];
    const float* ba1    = (const float*)d_in[6];
    const float* We1    = (const float*)d_in[7];
    const float* be1    = (const float*)d_in[8];
    const float* Wa2    = (const float*)d_in[9];
    const float* ba2    = (const float*)d_in[10];
    const float* We2    = (const float*)d_in[11];
    const float* be2    = (const float*)d_in[12];
    const float* Wbil   = (const float*)d_in[13];
    const float* bbil   = (const float*)d_in[14];

    // workspace carve-out (256B aligned)
    size_t off = 0;
    auto take = [&](size_t nbytes) -> float* {
        off = (off + 255) & ~(size_t)255;
        float* p = (float*)((char*)d_ws + off);
        off += nbytes;
        return p;
    };
    float* agg    = take((size_t)N * 128 * 4);   // 25.6 MB
    float* invdeg = take((size_t)N * 4);
    float* colsum = take(64 * 4);
    float* tvec   = take(64 * 4);
    float* acc    = take(4);
    float* hA     = take((size_t)N * 64 * 4);    // 12.8 MB
    float* hB     = take((size_t)N * 64 * 4);    // 12.8 MB
    float* bufX   = take((size_t)E * 64 * 4);    // 128 MB  (e1 scratch, then neg)
    float* bufY   = take((size_t)E * 64 * 4);    // 128 MB  (pos)

    const dim3 blk256(256);
    const int gN      = (N + 255) / 256;
    const int gE      = (E + 255) / 256;
    const int gAgg    = (N * 128 + 255) / 256;
    const int gScat   = (E * 32 + 255) / 256;
    const int gApply  = (N + 127) / 128;
    const int gEdge   = (E + 127) / 128;
    const float invE  = 1.0f / (float)E;

    // degrees -> 1/max(deg,1)
    zero_f32_kernel<<<gN, blk256, 0, stream>>>(invdeg, N);
    deg_count_kernel<<<gE, blk256, 0, stream>>>(dst, invdeg, E);
    invdeg_kernel<<<gN, blk256, 0, stream>>>(invdeg, N);
    zero_f32_kernel<<<1, 64, 0, stream>>>(colsum, 64);

    // ===== positive encode =====
    zero_f32_kernel<<<gAgg, blk256, 0, stream>>>(agg, N * 128);
    scatter_msgs_kernel<<<gScat, blk256, 0, stream>>>(nfeats, efeats, src, dst, perm, 0, agg, E);
    gemm_apply_kernel<<<gApply, blk256, 0, stream>>>(nfeats, agg, invdeg, Wa1, ba1, hA, N);
    gemm_edge_kernel<<<gEdge, blk256, 0, stream>>>(hA, src, dst, We1, be1, bufX, E);

    zero_f32_kernel<<<gAgg, blk256, 0, stream>>>(agg, N * 128);
    scatter_msgs_kernel<<<gScat, blk256, 0, stream>>>(hA, bufX, src, dst, perm, 0, agg, E);
    gemm_apply_kernel<<<gApply, blk256, 0, stream>>>(hA, agg, invdeg, Wa2, ba2, hB, N);
    gemm_edge_kernel<<<gEdge, blk256, 0, stream>>>(hB, src, dst, We2, be2, bufY, E);  // pos

    // ===== negative encode (permuted edge feats) =====
    zero_f32_kernel<<<gAgg, blk256, 0, stream>>>(agg, N * 128);
    scatter_msgs_kernel<<<gScat, blk256, 0, stream>>>(nfeats, efeats, src, dst, perm, 1, agg, E);
    gemm_apply_kernel<<<gApply, blk256, 0, stream>>>(nfeats, agg, invdeg, Wa1, ba1, hA, N);
    gemm_edge_kernel<<<gEdge, blk256, 0, stream>>>(hA, src, dst, We1, be1, bufX, E);

    zero_f32_kernel<<<gAgg, blk256, 0, stream>>>(agg, N * 128);
    scatter_msgs_kernel<<<gScat, blk256, 0, stream>>>(hA, bufX, src, dst, perm, 0, agg, E);
    gemm_apply_kernel<<<gApply, blk256, 0, stream>>>(hA, agg, invdeg, Wa2, ba2, hB, N);
    gemm_edge_kernel<<<gEdge, blk256, 0, stream>>>(hB, src, dst, We2, be2, bufX, E);  // neg (e1 dead)

    // ===== readout =====
    colsum_kernel<<<1024, blk256, 0, stream>>>(bufY, colsum, E);
    make_t_kernel<<<1, 64, 0, stream>>>(colsum, Wbil, tvec, acc, invE);
    loss_kernel<<<gE, blk256, 0, stream>>>(bufY, bufX, tvec, bbil, acc, E);
    finalize_kernel<<<1, 32, 0, stream>>>(acc, (float*)d_out, invE);
}
